// MACKRELRecurrentAgentLevel2_77515569758579
// MI455X (gfx1250) — matrix operand order repaired and checked by reading, not compiled, not run on hardware
//
#include <hip/hip_runtime.h>
#include <math.h>

typedef __attribute__((ext_vector_type(16))) __bf16 v16bf;
typedef __attribute__((ext_vector_type(8)))  float  v8f;
typedef __attribute__((ext_vector_type(4)))  unsigned int u32x4;
typedef __attribute__((ext_vector_type(4)))  float  f32x4;

#define TSTEPS   64
#define IN_DIMS  128
#define HID      64
#define OUT_DIM  146
#define OUT_PAD  160
#define NROW     3840          // 15 * 256 sequences
#define RPB      32            // rows per block = 2 waves x 16
#define TINYV    1.0842021724855044e-19f  // sqrt(FLT_MIN)

union BF16Frag { u32x4 q[2]; v16bf v; unsigned short s[16]; __bf16 h[16]; };

// hardware f32 -> bf16 convert (v_cvt_*bf16_f32) instead of bit-twiddled RNE
__device__ __forceinline__ unsigned short f2bf(float f) {
  __bf16 b = (__bf16)f;
  return __builtin_bit_cast(unsigned short, b);
}

// A-fragment (16x32 bf16, MxK): per-lane two contiguous 8-elem chunks at p, p+16
__device__ __forceinline__ v16bf afrag(const unsigned short* p) {
  BF16Frag u;
  u.q[0] = *(const u32x4*)(p);
  u.q[1] = *(const u32x4*)(p + 16);
  return u.v;
}
// B-fragment (32x16 bf16, KxN): per-lane two contiguous 8-elem K chunks at p, p+16 (row-major W[out][in])
__device__ __forceinline__ v16bf bfrag(const unsigned short* p) {
  BF16Frag u;
  u.q[0] = *(const u32x4*)(p);
  u.q[1] = *(const u32x4*)(p + 16);
  return u.v;
}
__device__ __forceinline__ v8f zero8() {
  v8f z = {0.f, 0.f, 0.f, 0.f, 0.f, 0.f, 0.f, 0.f};
  return z;
}
__device__ __forceinline__ float sigmoidf_(float x) { return 1.f / (1.f + __expf(-x)); }

#define WMMA_BF16(A, B, C) \
  __builtin_amdgcn_wmma_f32_16x16x32_bf16(false, (A), false, (B), (short)0, (C), false, false)

// ---- one-shot weight downcast fp32 -> bf16 (row-major [out][in]); Wo padded to 160 rows
__global__ void prep_weights(const float* __restrict__ W1, const float* __restrict__ Wih,
                             const float* __restrict__ Whh, const float* __restrict__ Wo,
                             unsigned short* __restrict__ W1b, unsigned short* __restrict__ Wihb,
                             unsigned short* __restrict__ Whhb, unsigned short* __restrict__ Wob) {
  int i = blockIdx.x * blockDim.x + threadIdx.x;
  if (i < HID * IN_DIMS) W1b[i] = f2bf(W1[i]);
  if (i < 3 * HID * HID) { Wihb[i] = f2bf(Wih[i]); Whhb[i] = f2bf(Whh[i]); }
  if (i < OUT_PAD * HID) {
    int r = i / HID;
    Wob[i] = (r < OUT_DIM) ? f2bf(Wo[i]) : (unsigned short)0;
  }
}

__launch_bounds__(64)
__global__ void mackrel_fused(const float* __restrict__ x, const float* __restrict__ h0,
                              const float* __restrict__ aa, const float* __restrict__ eps,
                              const unsigned short* __restrict__ W1b,
                              const unsigned short* __restrict__ Wihb,
                              const unsigned short* __restrict__ Whhb,
                              const unsigned short* __restrict__ Wob,
                              const float* __restrict__ b1, const float* __restrict__ bih,
                              const float* __restrict__ bhh, const float* __restrict__ bo,
                              float* __restrict__ out_x, float* __restrict__ out_h) {
  __shared__ __align__(16) unsigned short sXe[2][16][HID];   // encoder output, bf16
  __shared__ __align__(16) unsigned short sHb[2][16][HID];   // hidden, bf16 (WMMA A source)
  __shared__ float sHf[2][16][HID];                          // hidden, fp32 (exact recurrence)
  __shared__ float sO[2][16][OUT_PAD];                       // masked exp(logits)

  const int wave = threadIdx.x >> 5;
  const int lane = threadIdx.x & 31;
  const int l15  = lane & 15;
  const int hi   = lane >> 4;                 // half-wave select
  const int rowbase = blockIdx.x * RPB + wave * 16;

  // init hidden state for this wave's 16 rows
  for (int i = lane; i < 16 * HID; i += 32) {
    int r = i >> 6, c = i & 63;
    float v = h0[(size_t)(rowbase + r) * HID + c];
    sHf[wave][r][c] = v;
    sHb[wave][r][c] = f2bf(v);
  }
  __syncthreads();

  for (int t = 0; t < TSTEPS; ++t) {
    // ================= stage 1: encoder  Xe = relu(x_t @ W1^T + b1) =================
    v16bf ax[4];
    {
      const float* xp = x + ((size_t)(rowbase + l15) * TSTEPS + t) * IN_DIMS;
#pragma unroll
      for (int kt = 0; kt < 4; ++kt) {
        const int k0 = kt * 32 + hi * 8;
        f32x4 f0 = *(const f32x4*)(xp + k0);
        f32x4 f1 = *(const f32x4*)(xp + k0 + 4);
        f32x4 f2 = *(const f32x4*)(xp + k0 + 16);
        f32x4 f3 = *(const f32x4*)(xp + k0 + 20);
        BF16Frag u;
#pragma unroll
        for (int j = 0; j < 4; ++j) {
          u.h[j]      = (__bf16)f0[j];
          u.h[4 + j]  = (__bf16)f1[j];
          u.h[8 + j]  = (__bf16)f2[j];
          u.h[12 + j] = (__bf16)f3[j];
        }
        ax[kt] = u.v;
      }
      if (t + 1 < TSTEPS) __builtin_prefetch(xp + IN_DIMS, 0, 1);  // global_prefetch_b8
    }
#pragma unroll 1
    for (int nt = 0; nt < 4; ++nt) {
      v8f acc = zero8();
#pragma unroll
      for (int kt = 0; kt < 4; ++kt) {
        v16bf b = bfrag(W1b + (size_t)(nt * 16 + l15) * IN_DIMS + kt * 32 + hi * 8);
        acc = WMMA_BF16(ax[kt], b, acc);
      }
      const float bias = b1[nt * 16 + l15];
#pragma unroll
      for (int r = 0; r < 8; ++r) {
        float v = acc[r] + bias;
        v = v > 0.f ? v : 0.f;
        sXe[wave][hi ? r + 8 : r][nt * 16 + l15] = f2bf(v);
      }
    }
    __syncthreads();

    // ================= stage 2: fused GRU cell =================
    v16bf aXe[2], aHp[2];
#pragma unroll
    for (int kt = 0; kt < 2; ++kt) {
      aXe[kt] = afrag(&sXe[wave][l15][0] + kt * 32 + hi * 8);
      aHp[kt] = afrag(&sHb[wave][l15][0] + kt * 32 + hi * 8);
    }
#pragma unroll 1
    for (int nt = 0; nt < 4; ++nt) {
      v8f aR = zero8(), aZ = zero8(), aI = zero8(), aN = zero8();
#pragma unroll
      for (int kt = 0; kt < 2; ++kt) {
        const int kb = kt * 32 + hi * 8;
        const int cb = nt * 16 + l15;
        // r and z: Xe@W_ih + H@W_hh chained into one accumulator
        aR = WMMA_BF16(aXe[kt], bfrag(Wihb + (size_t)(0 * HID + cb) * HID + kb), aR);
        aR = WMMA_BF16(aHp[kt], bfrag(Whhb + (size_t)(0 * HID + cb) * HID + kb), aR);
        aZ = WMMA_BF16(aXe[kt], bfrag(Wihb + (size_t)(1 * HID + cb) * HID + kb), aZ);
        aZ = WMMA_BF16(aHp[kt], bfrag(Whhb + (size_t)(1 * HID + cb) * HID + kb), aZ);
        // n gate halves kept separate for r * h_n
        aI = WMMA_BF16(aXe[kt], bfrag(Wihb + (size_t)(2 * HID + cb) * HID + kb), aI);
        aN = WMMA_BF16(aHp[kt], bfrag(Whhb + (size_t)(2 * HID + cb) * HID + kb), aN);
      }
      const int col = nt * 16 + l15;
      const float br = bih[col] + bhh[col];
      const float bz = bih[HID + col] + bhh[HID + col];
      const float bi = bih[2 * HID + col];
      const float bn = bhh[2 * HID + col];
#pragma unroll
      for (int r = 0; r < 8; ++r) {
        const int row = hi ? r + 8 : r;
        const float rg = sigmoidf_(aR[r] + br);
        const float zg = sigmoidf_(aZ[r] + bz);
        const float ng = tanhf(aI[r] + bi + rg * (aN[r] + bn));
        const float hold = sHf[wave][row][col];
        const float hnew = (1.f - zg) * ng + zg * hold;
        sHf[wave][row][col] = hnew;
        sHb[wave][row][col] = f2bf(hnew);
        out_h[((size_t)(rowbase + row) * TSTEPS + t) * HID + col] = hnew;
      }
    }
    __syncthreads();

    // ================= stage 3: logits + masked exp =================
    v16bf aHn[2];
#pragma unroll
    for (int kt = 0; kt < 2; ++kt)
      aHn[kt] = afrag(&sHb[wave][l15][0] + kt * 32 + hi * 8);
#pragma unroll 1
    for (int nt = 0; nt < 10; ++nt) {
      v8f acc = zero8();
#pragma unroll
      for (int kt = 0; kt < 2; ++kt) {
        v16bf b = bfrag(Wob + (size_t)(nt * 16 + l15) * HID + kt * 32 + hi * 8);
        acc = WMMA_BF16(aHn[kt], b, acc);
      }
      const int col = nt * 16 + l15;
      if (col < OUT_DIM) {
        const float bias = bo[col];
#pragma unroll
        for (int r = 0; r < 8; ++r) {
          const int row = hi ? r + 8 : r;
          const float av = aa[((size_t)(rowbase + row) * TSTEPS + t) * OUT_DIM + col];
          const float o = acc[r] + bias;
          sO[wave][row][col] = (av == 0.f) ? TINYV : __expf(o);
        }
      }
    }
    __syncthreads();

    // ================= stage 4: normalization + epsilon exploration =================
    {
      const int row = l15;                       // 2 lanes per row
      const size_t n = (size_t)rowbase + row;
      const float* aap = aa + ((size_t)n * TSTEPS + t) * OUT_DIM;
      float* op = out_x + ((size_t)n * TSTEPS + t) * OUT_DIM;
      const int c0 = hi ? 73 : 0;
      const int c1 = hi ? OUT_DIM : 73;
      float s = 0.f, na = 0.f;
      for (int c = c0; c < c1; ++c) { na += aap[c]; s += sO[wave][row][c]; }
      s  += __shfl_xor(s, 16, 32);
      na += __shfl_xor(na, 16, 32);
      if (s <= TINYV * (float)OUT_DIM) s = 1.f;
      if (na == 1.f) na = 2.f;
      const float ep = eps[n];
      const float pscale = (1.f - ep) / s;
      const float escale = ep / (na - 1.f) * 0.5f;       // * (1 - DELEG_BIAS)
      for (int c = c0; c < c1; ++c) {
        const float p = sO[wave][row][c] * pscale;
        const float e = (c == 0) ? ep * 0.5f : aap[c] * escale;
        op[c] = e + p;
      }
    }
    __syncthreads();
  }
}

extern "C" void kernel_launch(void* const* d_in, const int* in_sizes, int n_in,
                              void* d_out, int out_size, void* d_ws, size_t ws_size,
                              hipStream_t stream) {
  const float* x   = (const float*)d_in[0];   // inputs_main [15,256,64,128]
  const float* h0  = (const float*)d_in[1];   // hidden_states [15,256,1,64]
  const float* aa  = (const float*)d_in[2];   // pairwise_avail_actions [15,256,64,146]
  const float* ep  = (const float*)d_in[3];   // epsilons [15,256,1]
  // d_in[4] sampled_pair_ids: unused by the reference math
  const float* W1  = (const float*)d_in[5];
  const float* b1  = (const float*)d_in[6];
  const float* Wih = (const float*)d_in[7];
  const float* bih = (const float*)d_in[8];
  const float* Whh = (const float*)d_in[9];
  const float* bhh = (const float*)d_in[10];
  const float* Wo  = (const float*)d_in[11];
  const float* bo  = (const float*)d_in[12];

  unsigned short* wsb  = (unsigned short*)d_ws;
  unsigned short* W1b  = wsb;                         // 64*128
  unsigned short* Wihb = W1b + HID * IN_DIMS;         // 192*64
  unsigned short* Whhb = Wihb + 3 * HID * HID;        // 192*64
  unsigned short* Wob  = Whhb + 3 * HID * HID;        // 160*64 (padded)

  float* out_x = (float*)d_out;                       // [15,256,64,146]
  float* out_h = out_x + (size_t)NROW * TSTEPS * OUT_DIM;  // [15,256,64,64]

  prep_weights<<<(3 * HID * HID + 255) / 256, 256, 0, stream>>>(
      W1, Wih, Whh, Wo, W1b, Wihb, Whhb, Wob);
  mackrel_fused<<<NROW / RPB, 64, 0, stream>>>(
      x, h0, aa, ep, W1b, Wihb, Whhb, Wob, b1, bih, bhh, bo, out_x, out_h);
}
